// StudentModel_20134806684074
// MI455X (gfx1250) — matrix operand order repaired and checked by reading, not compiled, not run on hardware
//
// CDNA5 / gfx1250 implementation of the dialogue-GNN reference.
// All dense GEMMs via v_wmma_f32_16x16x32_bf16 (bf16 in, f32 acc), 128x64 block
// tiles, vectorized b128 global->LDS staging (K padded to 32), contiguous-in-LDS
// fragments (2x ds_load_b128 each). Pairwise path-update / head matmuls are
// broadcast-factorized to avoid materializing the 268MB pair tensor.
#include <hip/hip_runtime.h>
#include <hip/hip_bf16.h>
#include <math.h>

typedef __bf16 bf16;
typedef __bf16 v16bf __attribute__((ext_vector_type(16)));
typedef float  v8f   __attribute__((ext_vector_type(8)));

// ---- model dims ----
#define GLOVE 300
#define GLOVEP 320     // padded K for WMMA tiles (zero-filled)
#define HID   512
#define H2    256
#define G3    768      // 3*H2
#define PH    128
#define DH    64
#define NREL  17
#define VDp   4
#define Bt    16
#define Et    63
#define Lt    32
#define Nt    64
#define BE    (Bt*Et)            // 1008
#define ROWS_S ((size_t)BE*Lt)   // 32256
#define NN    (Bt*Nt)            // 1024
#define PAIR  ((size_t)Bt*Nt*Nt) // 65536

static __device__ __forceinline__ float sigm(float x){ return 1.f/(1.f+__expf(-x)); }

// -------------------- utility kernels --------------------
__global__ void fill_f32(float* p, float v, size_t n){
  for (size_t i = (size_t)blockIdx.x*256 + threadIdx.x; i < n; i += (size_t)gridDim.x*256) p[i] = v;
}
__global__ void cvt_f32_bf16(const float* __restrict__ s, bf16* __restrict__ d, size_t n){
  for (size_t i = (size_t)blockIdx.x*256 + threadIdx.x; i < n; i += (size_t)gridDim.x*256) d[i] = (bf16)s[i];
}
// convert [R,K] f32 -> [R,Kp] bf16 with zero pad in columns K..Kp
__global__ void cvt_pad_bf16(const float* __restrict__ s, bf16* __restrict__ d, int R, int K, int Kp){
  size_t idx = (size_t)blockIdx.x*256 + threadIdx.x;
  size_t tot = (size_t)R*Kp;
  for (; idx < tot; idx += (size_t)gridDim.x*256){
    int r = (int)(idx / Kp), c = (int)(idx % Kp);
    d[idx] = (c < K) ? (bf16)s[(size_t)r*K + c] : (bf16)0.0f;
  }
}
// gather GloVe rows -> bf16 activations [ROWS_S, GLOVEP], pad zero
__global__ void gather_embed(const int* __restrict__ texts, const float* __restrict__ emb,
                             bf16* __restrict__ x){
  int rt = blockIdx.x;                 // (b*E + e)*L + t
  int tok = texts[rt];
  const float* e = emb + (size_t)tok*GLOVE;
  bf16* o = x + (size_t)rt*GLOVEP;
  for (int c = threadIdx.x; c < GLOVEP; c += blockDim.x)
    o[c] = (c < GLOVE) ? (bf16)e[c] : (bf16)0.0f;
}

// -------------------- WMMA GEMM --------------------
// C[M,N] = A[M,K](bf16) @ W[N,K](bf16)^T (+bias).
// Requirements: K % 32 == 0, lda % 8 == 0, ldw % 8 == 0, N % 64 == 0.
// Block tile 128x64, BK=32, 256 threads = 8 waves (4 in M x 2 in N),
// each wave computes 32x32 of C = 2x2 WMMA tiles (4 v_wmma per K step).
union Frag { uint4 u[2]; v16bf v; };

__global__ __launch_bounds__(256) void gemm_bf16_wmma(
    const bf16* __restrict__ A, int lda,
    const bf16* __restrict__ W, int ldw,
    float* __restrict__ C, int ldc,
    const float* __restrict__ bias,
    int M, int N, int K)
{
  __shared__ __attribute__((aligned(16))) bf16 As[128][32];
  __shared__ __attribute__((aligned(16))) bf16 Bst[64][32];   // [n][k]
  const int m0 = blockIdx.y*128, n0 = blockIdx.x*64;
  const int tid  = threadIdx.x;
  const int wave = tid >> 5, lane = tid & 31;
  const int wm = (wave & 3) * 32, wn = (wave >> 2) * 32;
  const int hh = lane >> 4, l = lane & 15;
  // per-thread staging coords (row-granular; K always in bounds after padding)
  const int ar = tid >> 1, ak = (tid & 1) * 16;   // A: 16 bf16 per thread
  const int br = tid >> 2, bk = (tid & 3) * 8;    // B: 8 bf16 per thread
  v8f acc[4];
#pragma unroll
  for (int i = 0; i < 4; ++i) acc[i] = (v8f){0,0,0,0,0,0,0,0};
  const int kt = K >> 5;
  for (int kb = 0; kb < kt; ++kb){
    const int k0 = kb * 32;
    // ---- stage A (128x32) : 2x b128 per thread, zero-fill rows >= M ----
    uint4 av0, av1;
    if (m0 + ar < M){
      const bf16* ap = A + (size_t)(m0 + ar)*lda + k0 + ak;
      av0 = *(const uint4*)(ap);
      av1 = *(const uint4*)(ap + 8);
    } else {
      av0.x = av0.y = av0.z = av0.w = 0u; av1 = av0;
    }
    *(uint4*)&As[ar][ak]     = av0;
    *(uint4*)&As[ar][ak + 8] = av1;
    // ---- stage B transposed (64 n x 32 k) : 1x b128 per thread ----
    *(uint4*)&Bst[br][bk] = *(const uint4*)(W + (size_t)(n0 + br)*ldw + k0 + bk);
    __syncthreads();
    if (kb + 1 < kt && m0 + ar < M)
      __builtin_prefetch(A + (size_t)(m0 + ar)*lda + k0 + 32 + ak, 0, 1);
    // ---- fragments (contiguous k order, same for A and B) ----
    // A: lanes 0-15 -> K 0..15, lanes 16-31 -> K 16..31; M = l
    Frag a0, a1, b0, b1;
    {
      const bf16* ap = &As[wm + l][0];
      a0.u[0] = *(const uint4*)(ap + hh*16);
      a0.u[1] = *(const uint4*)(ap + 8 + hh*16);
      const bf16* ap2 = &As[wm + 16 + l][0];
      a1.u[0] = *(const uint4*)(ap2 + hh*16);
      a1.u[1] = *(const uint4*)(ap2 + 8 + hh*16);
      // B: lanes 0-15 -> K 0..15, lanes 16-31 -> K 16..31; N = l
      const bf16* bp = &Bst[wn + l][hh*16];
      b0.u[0] = *(const uint4*)(bp);
      b0.u[1] = *(const uint4*)(bp + 8);
      const bf16* bp2 = &Bst[wn + 16 + l][hh*16];
      b1.u[0] = *(const uint4*)(bp2);
      b1.u[1] = *(const uint4*)(bp2 + 8);
    }
    acc[0] = __builtin_amdgcn_wmma_f32_16x16x32_bf16(false, a0.v, false, b0.v, (short)0, acc[0], false, false);
    acc[1] = __builtin_amdgcn_wmma_f32_16x16x32_bf16(false, a0.v, false, b1.v, (short)0, acc[1], false, false);
    acc[2] = __builtin_amdgcn_wmma_f32_16x16x32_bf16(false, a1.v, false, b0.v, (short)0, acc[2], false, false);
    acc[3] = __builtin_amdgcn_wmma_f32_16x16x32_bf16(false, a1.v, false, b1.v, (short)0, acc[3], false, false);
    __syncthreads();
  }
  // C layout: VGPR r -> M = r (lanes 0-15) / r+8 (lanes 16-31); N = lane&15
#pragma unroll
  for (int mi = 0; mi < 2; ++mi)
#pragma unroll
  for (int ni = 0; ni < 2; ++ni){
    v8f a = acc[mi*2 + ni];
    int col = n0 + wn + ni*16 + l;
    float bc = bias ? bias[col] : 0.f;
#pragma unroll
    for (int r = 0; r < 8; ++r){
      int row = m0 + wm + mi*16 + r + hh*8;
      if (row < M) C[(size_t)row*ldc + col] = a[r] + bc;
    }
  }
}

// -------------------- GRU gate (torch order: r,z,n) --------------------
__global__ void gru_gate(const float* __restrict__ gi, const float* __restrict__ gh,
                         float* __restrict__ h, float* __restrict__ ys,
                         int rows, int t, int T){
  int idx = blockIdx.x*blockDim.x + threadIdx.x;
  if (idx >= rows*H2) return;
  int r = idx / H2, c = idx - r*H2;
  const float* gir = gi + ((size_t)r*T + t)*G3;
  const float* ghr = gh + (size_t)r*G3;
  float rg = sigm(gir[c]        + ghr[c]);
  float zg = sigm(gir[H2 + c]   + ghr[H2 + c]);
  float ng = tanhf(gir[2*H2 + c] + rg*ghr[2*H2 + c]);
  float hn = (1.f - zg)*ng + zg*h[idx];
  h[idx] = hn;
  ys[((size_t)r*T + t)*H2 + c] = hn;
}

// -------------------- dialog-input assembly --------------------
__global__ void gather_fwd(const float* __restrict__ ys, const int* __restrict__ lengths,
                           float* __restrict__ din){
  int idx = blockIdx.x*blockDim.x + threadIdx.x;
  if (idx >= BE*H2) return;
  int c = idx % H2, be = idx / H2, b = be / Et, e = be % Et;
  int tl = lengths[be] - 1;
  din[((size_t)b*Nt + 1 + e)*HID + c] = ys[((size_t)be*Lt + tl)*H2 + c];
}
__global__ void gather_bwd(const float* __restrict__ ys, float* __restrict__ din){
  int idx = blockIdx.x*blockDim.x + threadIdx.x;
  if (idx >= BE*H2) return;
  int c = idx % H2, be = idx / H2, b = be / Et, e = be % Et;
  din[((size_t)b*Nt + 1 + e)*HID + H2 + c] = ys[((size_t)be*Lt + 0)*H2 + c];
}
__global__ void fill_root(const float* __restrict__ root, float* __restrict__ din){
  int idx = blockIdx.x*blockDim.x + threadIdx.x;
  if (idx >= Bt*HID) return;
  int b = idx / HID, c = idx % HID;
  din[(size_t)b*Nt*HID + c] = root[c];
}
__global__ void concat_fb(const float* __restrict__ df, const float* __restrict__ db,
                          float* __restrict__ o){
  int idx = blockIdx.x*blockDim.x + threadIdx.x;
  if (idx >= NN*H2) return;
  int row = idx / H2, c = idx % H2;
  o[(size_t)row*HID + c]      = df[idx];
  o[(size_t)row*HID + H2 + c] = db[idx];
}

// -------------------- layernorm (optionally x + [relu]res) --------------------
__global__ __launch_bounds__(128) void ln_kernel(const float* __restrict__ x,
                                                 const float* __restrict__ res, int relu,
                                                 const float* __restrict__ g, const float* __restrict__ b,
                                                 float* __restrict__ out, int C){
  size_t row = blockIdx.x;
  const float* xr = x + row*C;
  const float* rr = res ? res + row*C : nullptr;
  __shared__ float s1[128], s2[128];
  float a = 0.f, q = 0.f;
  for (int c = threadIdx.x; c < C; c += 128){
    float v = xr[c];
    if (rr){ float w = rr[c]; if (relu) w = fmaxf(w, 0.f); v += w; }
    a += v; q += v*v;
  }
  s1[threadIdx.x] = a; s2[threadIdx.x] = q; __syncthreads();
  for (int o = 64; o > 0; o >>= 1){
    if (threadIdx.x < o){ s1[threadIdx.x] += s1[threadIdx.x+o]; s2[threadIdx.x] += s2[threadIdx.x+o]; }
    __syncthreads();
  }
  float mean = s1[0] / C;
  float var  = s2[0] / C - mean*mean;
  float inv  = rsqrtf(var + 1e-5f);
  for (int c = threadIdx.x; c < C; c += 128){
    float v = xr[c];
    if (rr){ float w = rr[c]; if (relu) w = fmaxf(w, 0.f); v += w; }
    out[row*C + c] = (v - mean)*inv*g[c] + b[c];
  }
}

// -------------------- const path: [spk|turn|pos] --------------------
__global__ void build_const_path(const int* __restrict__ spk, const int* __restrict__ trn,
                                 const float* __restrict__ spk_emb, const float* __restrict__ turn_emb,
                                 const float* __restrict__ pos_emb, float* __restrict__ cp){
  size_t idx = (size_t)blockIdx.x*256 + threadIdx.x;
  if (idx >= PAIR*PH) return;
  int c = idx & (PH-1); size_t pr = idx >> 7;
  int j = pr & 63; size_t bi = pr >> 6; int i = bi & 63;
  float v;
  if (c < 32)       v = spk_emb[spk[pr]*32 + c];
  else if (c < 64)  v = turn_emb[trn[pr]*32 + (c - 32)];
  else {
    int d = j - i;
    d = d < -(VDp+1) ? -(VDp+1) : (d > (VDp+1) ? (VDp+1) : d);
    d += VDp + 1;
    v = pos_emb[d*64 + (c - 64)];
  }
  cp[idx] = v;
}

// -------------------- fused attention per (b, head) --------------------
__global__ __launch_bounds__(256) void attn_kernel(const float* __restrict__ q, const float* __restrict__ k,
                                                   const float* __restrict__ v,
                                                   const float* __restrict__ sk, const float* __restrict__ sv,
                                                   const int* __restrict__ edu_nums, float* __restrict__ out){
  __shared__ float qs[Nt][DH];
  __shared__ float ks[Nt][DH];
  __shared__ float sc[Nt][Nt];
  int b = blockIdx.x >> 3, h = blockIdx.x & 7;
  int tid = threadIdx.x;
  const size_t base = (size_t)b*Nt*HID + h*DH;
  for (int i = tid; i < Nt*DH; i += 256){
    int n = i >> 6, d = i & 63;
    qs[n][d] = q[base + (size_t)n*HID + d] * 0.125f;   // dh^-0.5
    ks[n][d] = k[base + (size_t)n*HID + d];
  }
  __syncthreads();
  int valid = edu_nums[b] + 1;
  const float* skb = sk + (size_t)b*Nt*Nt*DH;
  for (int i = tid; i < Nt*Nt; i += 256){
    int n = i >> 6, m = i & 63;
    const float* skr = skb + ((size_t)n*Nt + m)*DH;
    float s = 0.f;
    for (int d = 0; d < DH; ++d) s += qs[n][d]*(ks[m][d] + skr[d]);
    sc[n][m] = (m < valid) ? s : s - 1e9f;
  }
  __syncthreads();
  if (tid < Nt){
    float mx = -1e30f;
    for (int m = 0; m < Nt; ++m) mx = fmaxf(mx, sc[tid][m]);
    float ss = 0.f;
    for (int m = 0; m < Nt; ++m){ float e = __expf(sc[tid][m] - mx); sc[tid][m] = e; ss += e; }
    float r = 1.f/ss;
    for (int m = 0; m < Nt; ++m) sc[tid][m] *= r;
  }
  __syncthreads();
  const float* svb = sv + (size_t)b*Nt*Nt*DH;
  for (int i = tid; i < Nt*DH; i += 256){
    int n = i >> 6, d = i & 63;
    float o = 0.f;
    for (int m = 0; m < Nt; ++m)
      o += sc[n][m]*(v[base + (size_t)m*HID + d] + svb[((size_t)n*Nt + m)*DH + d]);
    out[base + (size_t)n*HID + d] = o;
  }
}

// -------------------- pairwise GRU path update (broadcast-factorized) --------------------
__global__ void path_update_apply(const float* __restrict__ Prj, const float* __restrict__ Pri,
                                  const float* __restrict__ Pzj, const float* __restrict__ Pzi,
                                  const float* __restrict__ Pcj, const float* __restrict__ Pci,
                                  const float* __restrict__ Hr,  const float* __restrict__ Hz,
                                  const float* __restrict__ Hu,
                                  const float* __restrict__ rb, const float* __restrict__ zb,
                                  const float* __restrict__ cb, const float* __restrict__ ub,
                                  float* __restrict__ hx, float* __restrict__ mem){
  size_t idx = (size_t)blockIdx.x*256 + threadIdx.x;
  if (idx >= PAIR*PH) return;
  int c = idx & (PH-1); size_t pr = idx >> 7;
  int j = pr & 63; size_t bi = pr >> 6; int i = bi & 63; int b = (int)(bi >> 6);
  size_t rj = ((size_t)b*Nt + j)*PH + c;   // cols 0:512 of W act on nodes[b,j]
  size_t ri = ((size_t)b*Nt + i)*PH + c;   // cols 512:1024 act on nodes[b,i]
  float r = sigm(Prj[rj] + Pri[ri] + Hr[idx] + rb[c]);
  float z = sigm(Pzj[rj] + Pzi[ri] + Hz[idx] + zb[c]);
  float u = tanhf(Pcj[rj] + Pci[ri] + cb[c] + r*(Hu[idx] + ub[c]));
  float h = hx[idx];
  float hn = z*h + (1.f - z)*u;
  hx[idx] = hn;
  mem[idx] = hn;
}

// -------------------- heads --------------------
__global__ __launch_bounds__(128) void link_head(const float* __restrict__ Ga, const float* __restrict__ Gb,
                                                 const float* __restrict__ b1, const float* __restrict__ W2,
                                                 const float* __restrict__ b2, float* __restrict__ link){
  size_t pr = blockIdx.x;
  int j = pr & 63; size_t bi = pr >> 6; int i = bi & 63; int b = (int)(bi >> 6);
  int c = threadIdx.x;
  size_t fij = pr*PH + c;
  size_t fji = (((size_t)b*Nt + j)*Nt + i)*PH + c;
  float t = tanhf(Ga[fij] + Gb[fji] + b1[c]);
  __shared__ float sh[128];
  sh[c] = t * W2[c];
  __syncthreads();
  for (int o = 64; o > 0; o >>= 1){ if (c < o) sh[c] += sh[c+o]; __syncthreads(); }
  if (c == 0) link[pr] = sh[0] + b2[0];
}
__global__ __launch_bounds__(128) void label_head(const float* __restrict__ Ga, const float* __restrict__ Gb,
                                                  const float* __restrict__ b1, const float* __restrict__ W2,
                                                  const float* __restrict__ b2, float* __restrict__ label){
  size_t pr = blockIdx.x;
  int j = pr & 63; size_t bi = pr >> 6; int i = bi & 63; int b = (int)(bi >> 6);
  int c = threadIdx.x;
  size_t fij = pr*PH + c;
  size_t fji = (((size_t)b*Nt + j)*Nt + i)*PH + c;
  __shared__ float sh[128];
  sh[c] = tanhf(Ga[fij] + Gb[fji] + b1[c]);
  __syncthreads();
  if (c < NREL){
    float s = 0.f;
    for (int d = 0; d < PH; ++d) s += sh[d]*W2[c*PH + d];
    label[pr*NREL + c] = s + b2[c];
  }
}

// =======================================================================
extern "C" void kernel_launch(void* const* d_in, const int* in_sizes, int n_in,
                              void* d_out, int out_size, void* d_ws, size_t ws_size,
                              hipStream_t stream) {
  // ---- inputs (setup_inputs order, then make_params leaves in insertion order) ----
  const int* texts   = (const int*)d_in[0];
  const int* lengths = (const int*)d_in[1];
  const int* edu     = (const int*)d_in[2];
  const int* spk     = (const int*)d_in[3];
  const int* trn     = (const int*)d_in[4];
  auto F = [&](int i)->const float*{ return (const float*)d_in[i]; };
  const float *p_emb = F(5), *p_root = F(6);
  const float *sgWi_f = F(7),  *sgWh_f = F(8),  *sgbi_f = F(9),  *sgbh_f = F(10);
  const float *sgWi_b = F(11), *sgWh_b = F(12), *sgbi_b = F(13), *sgbh_b = F(14);
  const float *dgWi_f = F(15), *dgWh_f = F(16), *dgbi_f = F(17), *dgbh_f = F(18);
  const float *dgWi_b = F(19), *dgWh_b = F(20), *dgbi_b = F(21), *dgbh_b = F(22);
  const float *spk_e = F(23), *turn_e = F(24), *pos_e = F(25);
  const float *qW = F(26), *qb = F(27), *kW = F(28), *kb_ = F(29);
  const float *vW = F(30), *vb = F(31), *oW = F(32), *ob = F(33);
  const float *skW = F(34), *skb_ = F(35), *svW = F(36), *svb_ = F(37);
  const float *png = F(38), *pnb = F(39), *ng_ = F(40), *nb_ = F(41), *normg = F(42), *normb = F(43);
  const float *purW = F(44), *purb = F(45), *puzW = F(46), *puzb = F(47);
  const float *pucW = F(48), *pucb = F(49), *puuW = F(50), *puub = F(51);
  const float *lkW1 = F(52), *lkb1 = F(53), *lkW2 = F(54), *lkb2 = F(55);
  const float *lbW1 = F(56), *lbb1 = F(57), *lbW2 = F(58), *lbb2 = F(59);

  float* out = (float*)d_out;
  float* out_link  = out;                      // [16,64,64]
  float* out_label = out + PAIR;               // [16,64,64,17]
  float* out_mem   = out + PAIR + PAIR*NREL;   // [2,16,64,64,128]

  // ---- workspace bump allocator ----
  char* ws = (char*)d_ws;
  size_t off = 0;
  auto A8 = [&](size_t bytes)->char*{ size_t o = off; off = (off + bytes + 255) & ~(size_t)255; return ws + o; };

  auto launch_fill = [&](float* p, float v, size_t n){
    unsigned g = (unsigned)(((n + 255)/256 < 16384) ? (n + 255)/256 : 16384);
    fill_f32<<<g, 256, 0, stream>>>(p, v, n);
  };
  auto launch_cvt = [&](const float* s, bf16* d, size_t n){
    unsigned g = (unsigned)(((n + 255)/256 < 16384) ? (n + 255)/256 : 16384);
    cvt_f32_bf16<<<g, 256, 0, stream>>>(s, d, n);
  };
  auto cvtW = [&](const float* s, size_t n)->bf16*{
    bf16* d = (bf16*)A8(n*sizeof(bf16)); launch_cvt(s, d, n); return d;
  };
  auto cvtWpad = [&](const float* s, int R, int K, int Kp)->bf16*{
    size_t n = (size_t)R*Kp;
    bf16* d = (bf16*)A8(n*sizeof(bf16));
    unsigned g = (unsigned)(((n + 255)/256 < 16384) ? (n + 255)/256 : 16384);
    cvt_pad_bf16<<<g, 256, 0, stream>>>(s, d, R, K, Kp);
    return d;
  };
  auto gemm = [&](const bf16* Ap, int lda, const bf16* Wp, int ldw,
                  float* Cp, int ldc, const float* bias, int M, int N, int K){
    dim3 grid((N + 63)/64, (M + 127)/128);
    gemm_bf16_wmma<<<grid, 256, 0, stream>>>(Ap, lda, Wp, ldw, Cp, ldc, bias, M, N, K);
  };

  // ---- persistent: bf16 weights (K padded where needed) ----
  bf16 *bWi_sf = cvtWpad(sgWi_f, G3, GLOVE, GLOVEP), *bWh_sf = cvtW(sgWh_f, (size_t)G3*H2);
  bf16 *bWi_sb = cvtWpad(sgWi_b, G3, GLOVE, GLOVEP), *bWh_sb = cvtW(sgWh_b, (size_t)G3*H2);
  bf16 *bWi_df = cvtW(dgWi_f, (size_t)G3*HID),       *bWh_df = cvtW(dgWh_f, (size_t)G3*H2);
  bf16 *bWi_db = cvtW(dgWi_b, (size_t)G3*HID),       *bWh_db = cvtW(dgWh_b, (size_t)G3*H2);
  bf16 *bqW = cvtW(qW, (size_t)HID*HID), *bkW = cvtW(kW, (size_t)HID*HID);
  bf16 *bvW = cvtW(vW, (size_t)HID*HID), *boW = cvtW(oW, (size_t)HID*HID);
  bf16 *bskW = cvtW(skW, (size_t)DH*PH), *bsvW = cvtW(svW, (size_t)DH*PH);
  bf16 *bpurW = cvtW(purW, (size_t)PH*1152), *bpuzW = cvtW(puzW, (size_t)PH*1152);
  bf16 *bpucW = cvtW(pucW, (size_t)PH*1024), *bpuuW = cvtW(puuW, (size_t)PH*PH);
  bf16 *blkW1 = cvtW(lkW1, (size_t)PH*256),  *blbW1 = cvtW(lbW1, (size_t)PH*256);

  // ---- persistent activations ----
  float* din    = (float*)A8((size_t)NN*HID*4);
  bf16*  din_bf = (bf16*) A8((size_t)NN*HID*2);
  float* nodes  = (float*)A8((size_t)NN*HID*4);
  bf16*  nodes_bf=(bf16*) A8((size_t)NN*HID*2);
  float* cpath  = (float*)A8(PAIR*PH*4);
  float* hx     = (float*)A8(PAIR*PH*4);
  bf16*  hx_bf  = (bf16*) A8(PAIR*PH*2);
  float* ys_df  = (float*)A8((size_t)NN*H2*4);
  float* ys_db  = (float*)A8((size_t)NN*H2*4);
  float* d_go   = (float*)A8((size_t)NN*HID*4);   // dialog_output
  size_t scratch0 = off;

  // =================== phase 1: sentence BiGRU ===================
  off = scratch0;
  bf16*  x_bf = (bf16*) A8(ROWS_S*GLOVEP*2);
  float* gi   = (float*)A8(ROWS_S*G3*4);
  float* ysS  = (float*)A8(ROWS_S*H2*4);
  float* h    = (float*)A8((size_t)BE*H2*4);
  bf16*  h_bf = (bf16*) A8((size_t)BE*H2*2);
  float* gh   = (float*)A8((size_t)BE*G3*4);
  float* gid  = (float*)A8((size_t)NN*G3*4);

  gather_embed<<<(unsigned)ROWS_S, 128, 0, stream>>>(texts, p_emb, x_bf);

  // forward direction
  gemm(x_bf, GLOVEP, bWi_sf, GLOVEP, gi, G3, sgbi_f, (int)ROWS_S, G3, GLOVEP);
  launch_fill(h, 0.f, (size_t)BE*H2);
  for (int t = 0; t < Lt; ++t){
    launch_cvt(h, h_bf, (size_t)BE*H2);
    gemm(h_bf, H2, bWh_sf, H2, gh, G3, sgbh_f, BE, G3, H2);
    gru_gate<<<(BE*H2 + 255)/256, 256, 0, stream>>>(gi, gh, h, ysS, BE, t, Lt);
  }
  gather_fwd<<<(BE*H2 + 255)/256, 256, 0, stream>>>(ysS, lengths, din);

  // backward direction
  gemm(x_bf, GLOVEP, bWi_sb, GLOVEP, gi, G3, sgbi_b, (int)ROWS_S, G3, GLOVEP);
  launch_fill(h, 0.f, (size_t)BE*H2);
  for (int t = Lt - 1; t >= 0; --t){
    launch_cvt(h, h_bf, (size_t)BE*H2);
    gemm(h_bf, H2, bWh_sb, H2, gh, G3, sgbh_b, BE, G3, H2);
    gru_gate<<<(BE*H2 + 255)/256, 256, 0, stream>>>(gi, gh, h, ysS, BE, t, Lt);
  }
  gather_bwd<<<(BE*H2 + 255)/256, 256, 0, stream>>>(ysS, din);
  fill_root<<<(Bt*HID + 255)/256, 256, 0, stream>>>(p_root, din);

  // =================== phase 2: dialog BiGRU ===================
  launch_cvt(din, din_bf, (size_t)NN*HID);
  gemm(din_bf, HID, bWi_df, HID, gid, G3, dgbi_f, NN, G3, HID);
  launch_fill(h, 0.f, (size_t)Bt*H2);
  for (int t = 0; t < Nt; ++t){
    launch_cvt(h, h_bf, (size_t)Bt*H2);
    gemm(h_bf, H2, bWh_df, H2, gh, G3, dgbh_f, Bt, G3, H2);
    gru_gate<<<(Bt*H2 + 255)/256, 256, 0, stream>>>(gid, gh, h, ys_df, Bt, t, Nt);
  }
  gemm(din_bf, HID, bWi_db, HID, gid, G3, dgbi_b, NN, G3, HID);
  launch_fill(h, 0.f, (size_t)Bt*H2);
  for (int t = Nt - 1; t >= 0; --t){
    launch_cvt(h, h_bf, (size_t)Bt*H2);
    gemm(h_bf, H2, bWh_db, H2, gh, G3, dgbh_b, Bt, G3, H2);
    gru_gate<<<(Bt*H2 + 255)/256, 256, 0, stream>>>(gid, gh, h, ys_db, Bt, t, Nt);
  }
  concat_fb<<<(NN*H2 + 255)/256, 256, 0, stream>>>(ys_df, ys_db, d_go);
  ln_kernel<<<NN, 128, 0, stream>>>(din, d_go, 0, normg, normb, nodes, HID);

  // const path + zero struct path
  build_const_path<<<(unsigned)((PAIR*PH + 255)/256), 256, 0, stream>>>(spk, trn, spk_e, turn_e, pos_e, cpath);
  launch_fill(hx, 0.f, PAIR*PH);

  // =================== phase 3: GNN layers ===================
  off = scratch0;
  float* pn    = (float*)A8(PAIR*PH*4);
  bf16*  pn_bf = (bf16*) A8(PAIR*PH*2);
  float* skB   = (float*)A8(PAIR*DH*4);
  float* svB   = (float*)A8(PAIR*DH*4);
  float* qB    = (float*)A8((size_t)NN*HID*4);
  float* kB    = (float*)A8((size_t)NN*HID*4);
  float* vB    = (float*)A8((size_t)NN*HID*4);
  float* aoB   = (float*)A8((size_t)NN*HID*4);
  bf16*  ao_bf = (bf16*) A8((size_t)NN*HID*2);
  float* oprj  = (float*)A8((size_t)NN*HID*4);
  float* Prj = (float*)A8((size_t)NN*PH*4);
  float* Pri = (float*)A8((size_t)NN*PH*4);
  float* Pzj = (float*)A8((size_t)NN*PH*4);
  float* Pzi = (float*)A8((size_t)NN*PH*4);
  float* Pcj = (float*)A8((size_t)NN*PH*4);
  float* Pci = (float*)A8((size_t)NN*PH*4);
  float* Hr  = (float*)A8(PAIR*PH*4);
  float* Hz  = (float*)A8(PAIR*PH*4);
  float* Hu  = (float*)A8(PAIR*PH*4);

  for (int l = 0; l < 2; ++l){
    // paths = LN(struct + const)
    ln_kernel<<<(unsigned)PAIR, 128, 0, stream>>>(hx, cpath, 0, png, pnb, pn, PH);
    launch_cvt(pn, pn_bf, PAIR*PH);
    launch_cvt(nodes, nodes_bf, (size_t)NN*HID);
    gemm(nodes_bf, HID, bqW, HID, qB, HID, qb, NN, HID, HID);
    gemm(nodes_bf, HID, bkW, HID, kB, HID, kb_, NN, HID, HID);
    gemm(nodes_bf, HID, bvW, HID, vB, HID, vb, NN, HID, HID);
    gemm(pn_bf, PH, bskW, PH, skB, DH, skb_, (int)PAIR, DH, PH);
    gemm(pn_bf, PH, bsvW, PH, svB, DH, svb_, (int)PAIR, DH, PH);
    attn_kernel<<<Bt*8, 256, 0, stream>>>(qB, kB, vB, skB, svB, edu, aoB);
    launch_cvt(aoB, ao_bf, (size_t)NN*HID);
    gemm(ao_bf, HID, boW, HID, oprj, HID, ob, NN, HID, HID);
    ln_kernel<<<NN, 128, 0, stream>>>(nodes, oprj, 1, ng_, nb_, nodes, HID);

    // path update (factorized):  W[:, :512] -> nodes[b,j],  W[:,512:1024] -> nodes[b,i],  W[:,1024:] -> hx
    launch_cvt(nodes, nodes_bf, (size_t)NN*HID);
    gemm(nodes_bf, HID, bpurW,        1152, Prj, PH, nullptr, NN, PH, HID);
    gemm(nodes_bf, HID, bpurW + 512,  1152, Pri, PH, nullptr, NN, PH, HID);
    gemm(nodes_bf, HID, bpuzW,        1152, Pzj, PH, nullptr, NN, PH, HID);
    gemm(nodes_bf, HID, bpuzW + 512,  1152, Pzi, PH, nullptr, NN, PH, HID);
    gemm(nodes_bf, HID, bpucW,        1024, Pcj, PH, nullptr, NN, PH, HID);
    gemm(nodes_bf, HID, bpucW + 512,  1024, Pci, PH, nullptr, NN, PH, HID);
    launch_cvt(hx, hx_bf, PAIR*PH);
    gemm(hx_bf, PH, bpurW + 1024, 1152, Hr, PH, nullptr, (int)PAIR, PH, PH);
    gemm(hx_bf, PH, bpuzW + 1024, 1152, Hz, PH, nullptr, (int)PAIR, PH, PH);
    gemm(hx_bf, PH, bpuuW,        PH,   Hu, PH, nullptr, (int)PAIR, PH, PH);
    path_update_apply<<<(unsigned)((PAIR*PH + 255)/256), 256, 0, stream>>>(
        Prj, Pri, Pzj, Pzi, Pcj, Pci, Hr, Hz, Hu,
        purb, puzb, pucb, puub, hx, out_mem + (size_t)l*PAIR*PH);
  }

  // =================== phase 4: heads ===================
  // pred = [hx(i,j) | hx(j,i)]; W1 split into [:, :128] and [:, 128:]
  launch_cvt(hx, hx_bf, PAIR*PH);
  gemm(hx_bf, PH, blkW1,       256, Hr, PH, nullptr, (int)PAIR, PH, PH);
  gemm(hx_bf, PH, blkW1 + 128, 256, Hz, PH, nullptr, (int)PAIR, PH, PH);
  link_head<<<(unsigned)PAIR, 128, 0, stream>>>(Hr, Hz, lkb1, lkW2, lkb2, out_link);
  gemm(hx_bf, PH, blbW1,       256, Hr, PH, nullptr, (int)PAIR, PH, PH);
  gemm(hx_bf, PH, blbW1 + 128, 256, Hz, PH, nullptr, (int)PAIR, PH, PH);
  label_head<<<(unsigned)PAIR, 128, 0, stream>>>(Hr, Hz, lbb1, lbW2, lbb2, out_label);

  (void)in_sizes; (void)n_in; (void)out_size; (void)ws_size;
}
